// TBCNNFFDLayer_39367670235354
// MI455X (gfx1250) — matrix operand order repaired and checked by reading, not compiled, not run on hardware
//
#include <hip/hip_runtime.h>

typedef __attribute__((ext_vector_type(16))) __bf16 v16bf;
typedef __attribute__((ext_vector_type(8)))  float  v8f;

#define BATCH 512
#define NN    64
#define CC    8
#define DD    256
#define KTOT  768          // 3*D (t | l | r blocks)
#define APAD  776          // 768 + 8 pad: row stride = 1552B -> 4-bank row skew, b128 reads tile banks
#define NEGV  (-1e8f)

__device__ __forceinline__ unsigned short f2bf(float f) {
  unsigned u = __float_as_uint(f);
  u += 0x7fffu + ((u >> 16) & 1u);   // round-to-nearest-even
  return (unsigned short)(u >> 16);
}

union FragBF { v16bf v; uint4 q[2]; };

// Stage Wcat^T in bf16 into d_ws: Wtb[o][k] = w_e[d][o], e=k/256, d=k%256.
// B-fragment for WMMA then reads 32 contiguous bytes per lane.
__global__ void __launch_bounds__(256) prep_weights_kernel(
    const float* __restrict__ wt, const float* __restrict__ wl,
    const float* __restrict__ wr, unsigned short* __restrict__ Wtb) {
  int k = blockIdx.x;      // 0..767
  int o = threadIdx.x;     // 0..255 (coalesced read along o)
  int e = k >> 8, d = k & 255;
  const float* W = (e == 0) ? wt : ((e == 1) ? wl : wr);
  Wtb[(size_t)o * KTOT + k] = f2bf(W[d * DD + o]);
}

__global__ void __launch_bounds__(256) tbcnn_kernel(
    const float* __restrict__ x, const int* __restrict__ childIdx,
    const int* __restrict__ treeMask, const float* __restrict__ bias,
    const float* __restrict__ gamma, const float* __restrict__ beta,
    const float* __restrict__ attnW, const float* __restrict__ attnB,
    const float* __restrict__ outW, const float* __restrict__ outB,
    const unsigned short* __restrict__ Wtb, float* __restrict__ out) {
  __shared__ unsigned short As[NN * APAD];   // 99,328 B : [S_t|S_l|S_r] bf16
  __shared__ float hs[NN * DD];              // 65,536 B : conv, then relu(LN(h))
  __shared__ float logitsS[NN];
  __shared__ float scoresS[NN];
  __shared__ float pooledS[DD];

  const int b    = blockIdx.x;
  const int tid  = threadIdx.x;   // 256 threads = 8 wave32
  const int wv   = tid >> 5;
  const int lane = tid & 31;
  const float* xb = x + (size_t)b * NN * DD;

  // ---- Phase 1: eta-weighted child sums -> A[64][768] bf16 in LDS ----
  for (int n = 0; n < NN; ++n) {
    const int* ch = childIdx + ((size_t)b * NN + n) * CC;  // uniform -> scalarized
    int cv[8];
    #pragma unroll
    for (int c = 0; c < 8; ++c) cv[c] = ch[c];
    int ns = 0;
    #pragma unroll
    for (int c = 0; c < 8; ++c) ns += (cv[c] != 0);
    float den = (ns > 1) ? (float)(ns - 1) : 1.0f;
    float at = xb[n * DD + tid];
    float al = 0.f, ar = 0.f;
    #pragma unroll
    for (int c = 0; c < 8; ++c) {
      if (cv[c] != 0) {                    // lookup[0] == 0 => zero contribution
        float er = (ns == 1) ? ((c == 0) ? 0.5f : 0.0f) : ((float)c / den);
        float el = 1.0f - er;
        float v  = xb[cv[c] * DD + tid];   // coalesced row gather
        al = fmaf(el, v, al);
        ar = fmaf(er, v, ar);
      }
    }
    As[n * APAD + tid]       = f2bf(at);
    As[n * APAD + 256 + tid] = f2bf(al);
    As[n * APAD + 512 + tid] = f2bf(ar);
  }
  __syncthreads();

  // ---- Phase 2: conv = A @ Wcat  (M=64, K=768, N=256), bf16 WMMA ----
  {
    const int half = lane >> 4;     // K-half select: lane half h holds K=16h..16h+15
    const int lr   = lane & 15;
    #pragma unroll
    for (int i = 0; i < 8; ++i) {
      int t  = wv * 8 + i;          // 64 tiles / 8 waves
      int mt = t & 3, nt = t >> 2;
      // A lane row = mt*16+lr ; elems j -> K=16h+j (contiguous)
      const unsigned short* Arow = As + (mt * 16 + lr) * APAD + half * 16;
      // B lane col = nt*16+lr ; elems j -> K=16h+j (contiguous 32B)
      const unsigned short* Brow = Wtb + (size_t)(nt * 16 + lr) * KTOT + half * 16;
      v8f acc = {0.f, 0.f, 0.f, 0.f, 0.f, 0.f, 0.f, 0.f};
      for (int kt = 0; kt < KTOT / 32; ++kt) {
        FragBF a, bb;
        a.q[0] = *(const uint4*)(Arow + kt * 32);        // ds_load_b128
        a.q[1] = *(const uint4*)(Arow + kt * 32 + 8);    // ds_load_b128
        const uint4* pb = (const uint4*)(Brow + kt * 32);
        bb.q[0] = pb[0];                                 // global_load_b128 (L2 hot)
        bb.q[1] = pb[1];
        acc = __builtin_amdgcn_wmma_f32_16x16x32_bf16(
            false, a.v, false, bb.v, (short)0, acc, false, false);
      }
      #pragma unroll
      for (int r = 0; r < 8; ++r)   // D elem r -> row mt*16 + r + 8*half, col nt*16+lr
        hs[(mt * 16 + r + 8 * half) * DD + nt * 16 + lr] = acc[r];
    }
  }
  __syncthreads();

  // ---- Phase 3: residual + LayerNorm + ReLU + attention logit (wave per row) ----
  for (int j = 0; j < 8; ++j) {
    int n = wv * 8 + j;
    float v[8];
    float s = 0.f, s2 = 0.f;
    #pragma unroll
    for (int i = 0; i < 8; ++i) {
      int d = lane + 32 * i;
      float h = hs[n * DD + d] + bias[d] + xb[n * DD + d];
      v[i] = h; s += h; s2 = fmaf(h, h, s2);
    }
    #pragma unroll
    for (int off = 16; off > 0; off >>= 1) {
      s  += __shfl_xor(s,  off, 32);
      s2 += __shfl_xor(s2, off, 32);
    }
    float mu  = s * (1.0f / 256.0f);
    float var = s2 * (1.0f / 256.0f) - mu * mu;
    float inv = rsqrtf(var + 1e-5f);
    float lg = 0.f;
    #pragma unroll
    for (int i = 0; i < 8; ++i) {
      int d = lane + 32 * i;
      float h = (v[i] - mu) * inv * gamma[d] + beta[d];
      h = fmaxf(h, 0.f);
      hs[n * DD + d] = h;
      lg = fmaf(h, attnW[d], lg);
    }
    #pragma unroll
    for (int off = 16; off > 0; off >>= 1) lg += __shfl_xor(lg, off, 32);
    if (lane == 0) {
      float madd = (treeMask[(size_t)b * NN + n] != 0) ? NEGV : 0.0f;
      logitsS[n] = lg + attnB[0] + madd;
    }
  }
  __syncthreads();

  // ---- Phase 4: softmax over 64 nodes (wave 0) ----
  if (wv == 0) {
    float l0 = logitsS[lane], l1 = logitsS[lane + 32];
    float mx = fmaxf(l0, l1);
    #pragma unroll
    for (int off = 16; off > 0; off >>= 1) mx = fmaxf(mx, __shfl_xor(mx, off, 32));
    float e0 = __expf(l0 - mx), e1 = __expf(l1 - mx);
    float ss = e0 + e1;
    #pragma unroll
    for (int off = 16; off > 0; off >>= 1) ss += __shfl_xor(ss, off, 32);
    float r = 1.0f / ss;
    scoresS[lane]      = e0 * r;
    scoresS[lane + 32] = e1 * r;
  }
  __syncthreads();

  // pooled = scores^T @ h  (scores sum to 1, so out_b commutes through pooling)
  float p = 0.f;
  for (int n = 0; n < NN; ++n) p = fmaf(scoresS[n], hs[n * DD + tid], p);
  pooledS[tid] = p;
  __syncthreads();

  // ---- Phase 5: out[b] = pooled @ out_w + out_b (mat-vec, coalesced outW rows) ----
  float acc = outB[tid];
  for (int k = 0; k < DD; ++k) acc = fmaf(pooledS[k], outW[k * DD + tid], acc);
  out[(size_t)b * DD + tid] = acc;
}

extern "C" void kernel_launch(void* const* d_in, const int* in_sizes, int n_in,
                              void* d_out, int out_size, void* d_ws, size_t ws_size,
                              hipStream_t stream) {
  (void)in_sizes; (void)n_in; (void)out_size; (void)ws_size;
  const float* x     = (const float*)d_in[0];
  const int*   child = (const int*)d_in[1];
  const int*   tmask = (const int*)d_in[2];
  const float* wt    = (const float*)d_in[3];
  const float* wl    = (const float*)d_in[4];
  const float* wr    = (const float*)d_in[5];
  const float* bias  = (const float*)d_in[6];
  const float* gam   = (const float*)d_in[7];
  const float* bet   = (const float*)d_in[8];
  const float* attnW = (const float*)d_in[9];
  const float* attnB = (const float*)d_in[10];
  const float* outW  = (const float*)d_in[11];
  const float* outB  = (const float*)d_in[12];
  float* out = (float*)d_out;
  unsigned short* Wtb = (unsigned short*)d_ws;   // 768*256*2 = 384 KB scratch

  prep_weights_kernel<<<KTOT, 256, 0, stream>>>(wt, wl, wr, Wtb);
  tbcnn_kernel<<<BATCH, 256, 0, stream>>>(x, child, tmask, bias, gam, bet,
                                          attnW, attnB, outW, outB, Wtb, out);
}
